// Attention_70944269796047
// MI455X (gfx1250) — 16-bit and 8-bit matrix operand order repaired and checked by reading, compiled, not run on hardware
//
#include <hip/hip_runtime.h>

#define B_   2
#define S_   1024
#define H_   2048
#define NH_  16
#define NKV_ 8
#define HD_  128
#define NREP_ 2
#define EPS_ 1e-6f

typedef __attribute__((ext_vector_type(16))) _Float16 v16h;
typedef __attribute__((ext_vector_type(8)))  float    v8f;
typedef __attribute__((ext_vector_type(8)))  int      v8i;
typedef __attribute__((ext_vector_type(4)))  int      v4i;

union FragH { v16h v; unsigned int u[8]; };
union FragI { v8i  v; unsigned int u[8]; };
union Pack8H { int4 v; _Float16 h[8]; };

#if defined(__HIP_DEVICE_COMPILE__) && __has_builtin(__builtin_amdgcn_global_load_async_to_lds_b128)
#define ASYNC_OK 1
#else
#define ASYNC_OK 0
#endif

#if ASYNC_OK
typedef __attribute__((address_space(1))) v4i GB128;
typedef __attribute__((address_space(3))) v4i LB128;
#if __has_builtin(__builtin_amdgcn_s_wait_asynccnt)
#define ASYNC_WAIT(n) __builtin_amdgcn_s_wait_asynccnt(n)
#else
#define ASYNC_WAIT(n) asm volatile("s_wait_asynccnt %0" :: "n"(n) : "memory")
#endif
#else
#define ASYNC_WAIT(n) ((void)0)
#endif

__device__ __forceinline__ void async_copy16(const void* g, void* l) {
#if ASYNC_OK
    __builtin_amdgcn_global_load_async_to_lds_b128((GB128*)g, (LB128*)l, 0, 0);
#else
    *(int4*)l = *(const int4*)g;
#endif
}

__global__ void k_abs_partial(const float* __restrict__ w, int n, float* __restrict__ part) {
    __shared__ float red[256];
    int tid = threadIdx.x;
    float s = 0.f;
    for (int i = blockIdx.x * 256 + tid; i < n; i += gridDim.x * 256)
        s += fabsf(w[i]);
    red[tid] = s;
    __syncthreads();
    for (int st = 128; st > 0; st >>= 1) {
        if (tid < st) red[tid] += red[tid + st];
        __syncthreads();
    }
    if (tid == 0) part[blockIdx.x] = red[0];
}

__global__ void k_abs_finish(const float* __restrict__ part, float n, float* __restrict__ sc) {
    __shared__ float red[256];
    int tid = threadIdx.x;
    red[tid] = part[tid];
    __syncthreads();
    for (int st = 128; st > 0; st >>= 1) {
        if (tid < st) red[tid] += red[tid + st];
        __syncthreads();
    }
    if (tid == 0) sc[0] = fmaxf(red[0] / n, 1e-5f);
}

__global__ void k_wquant(const float* __restrict__ w, const float* __restrict__ sc,
                         signed char* __restrict__ wq, int n) {
    float inv = 1.f / sc[0];
    for (int i = blockIdx.x * blockDim.x + threadIdx.x; i < n; i += gridDim.x * blockDim.x) {
        float t = rintf(w[i] * inv);
        t = fminf(1.f, fmaxf(-1.f, t));
        wq[i] = (signed char)t;
    }
}

__global__ void k_actquant(const float* __restrict__ x, signed char* __restrict__ xq,
                           float* __restrict__ rs, int K) {
    __shared__ float red[256];
    int row = blockIdx.x, tid = threadIdx.x;
    const float* xr = x + (size_t)row * K;
    float am = 0.f;
    for (int i = tid; i < K; i += 256) am = fmaxf(am, fabsf(xr[i]));
    red[tid] = am;
    __syncthreads();
    for (int st = 128; st > 0; st >>= 1) {
        if (tid < st) red[tid] = fmaxf(red[tid], red[tid + st]);
        __syncthreads();
    }
    float amax = fmaxf(red[0], 1e-5f);
    float s = 127.f / amax;
    signed char* qr = xq + (size_t)row * K;
    for (int i = tid; i < K; i += 256) {
        float q = rintf(xr[i] * s);
        q = fminf(127.f, fmaxf(-128.f, q));
        qr[i] = (signed char)q;
    }
    if (tid == 0) rs[row] = amax / 127.f;
}

__global__ __launch_bounds__(256) void k_gemm_i8_pipe(
        const signed char* __restrict__ A, const float* __restrict__ as,
        const signed char* __restrict__ W, const float* __restrict__ wsc,
        float* __restrict__ C, int M, int N, int K) {
    __shared__ signed char sA[2][64 * 64];
    __shared__ signed char sB[2][64 * 64];
    int tid = threadIdx.x, lane = tid & 31, wave = tid >> 5;
    int g = lane >> 4, ln = lane & 15;
    int m0 = blockIdx.y * 64, n0 = blockIdx.x * 64;
    int cR = (wave >> 1) * 16, cC = (wave & 1) * 32;
    int lr = tid >> 2, lcb = (tid & 3) * 16;
    const signed char* ga = &A[(size_t)(m0 + lr) * K + lcb];
    const signed char* gb = &W[(size_t)(n0 + lr) * K + lcb];
    int loff = lr * 64 + lcb;

    v8i acc0 = {}, acc1 = {};
    int nsteps = K >> 6;
    async_copy16(ga, &sA[0][loff]);
    async_copy16(gb, &sB[0][loff]);
    for (int s2 = 0; s2 < nsteps; ++s2) {
        int cur = s2 & 1;
        if (s2 + 1 < nsteps) {
            async_copy16(ga + (size_t)(s2 + 1) * 64, &sA[cur ^ 1][loff]);
            async_copy16(gb + (size_t)(s2 + 1) * 64, &sB[cur ^ 1][loff]);
            ASYNC_WAIT(2);
        } else {
            ASYNC_WAIT(0);
        }
        __syncthreads();
        FragI a, b0, b1;
#pragma unroll
        for (int j = 0; j < 8; ++j) {
            int ka = (j >> 2) * 32 + g * 16 + (j & 3) * 4;
            a.u[j] = *(const unsigned int*)&sA[cur][(cR + ln) * 64 + ka];
            int kb = (j >> 2) * 32 + g * 16 + (j & 3) * 4;
            b0.u[j] = *(const unsigned int*)&sB[cur][(cC + ln) * 64 + kb];
            b1.u[j] = *(const unsigned int*)&sB[cur][(cC + 16 + ln) * 64 + kb];
        }
        acc0 = __builtin_amdgcn_wmma_i32_16x16x64_iu8(true, a.v, true, b0.v, acc0, false, false);
        acc1 = __builtin_amdgcn_wmma_i32_16x16x64_iu8(true, a.v, true, b1.v, acc1, false, false);
        __syncthreads();
    }
    float wf = wsc[0];
#pragma unroll
    for (int r = 0; r < 8; ++r) {
        int m = m0 + cR + r + g * 8;
        int n = n0 + cC + ln;
        float dq = as[m] * wf;
        C[(size_t)m * N + n]      = (float)acc0[r] * dq;
        C[(size_t)m * N + n + 16] = (float)acc1[r] * dq;
    }
}

__global__ void k_rmsrope(const float* __restrict__ X, const float* __restrict__ gamma,
                          const float* __restrict__ cosT, const float* __restrict__ sinT,
                          _Float16* __restrict__ out, int nH) {
    __shared__ float sq[HD_];
    __shared__ float nv[HD_];
    int idx = blockIdx.x;
    int h = idx % nH, m = idx / nH;
    int s = m % S_, b = m / S_;
    int d = threadIdx.x;
    float v = X[(size_t)m * nH * HD_ + h * HD_ + d];
    sq[d] = v * v;
    __syncthreads();
    for (int st = 64; st > 0; st >>= 1) {
        if (d < st) sq[d] += sq[d + st];
        __syncthreads();
    }
    float rms = rsqrtf(sq[0] / (float)HD_ + EPS_);
    nv[d] = gamma[d] * v * rms;
    __syncthreads();
    float rot = (d < 64) ? -nv[d + 64] : nv[d - 64];
    float o = nv[d] * cosT[s * HD_ + d] + rot * sinT[s * HD_ + d];
    out[(((size_t)b * nH + h) * S_ + s) * HD_ + d] = (_Float16)o;
}

__global__ void k_vconv(const float* __restrict__ Vf, _Float16* __restrict__ vh) {
    int n = B_ * S_ * NKV_ * HD_;
    for (int i = blockIdx.x * blockDim.x + threadIdx.x; i < n; i += gridDim.x * blockDim.x) {
        int d = i & (HD_ - 1);
        int hh = (i >> 7) & (NKV_ - 1);
        int m = i >> 10;
        int s = m & (S_ - 1);
        int b = m >> 10;
        vh[(((size_t)b * NKV_ + hh) * S_ + s) * HD_ + d] = (_Float16)Vf[i];
    }
}

__global__ __launch_bounds__(128) void k_attn(const _Float16* __restrict__ Q,
                                              const _Float16* __restrict__ K,
                                              const _Float16* __restrict__ V,
                                              float* __restrict__ AO) {
    __shared__ _Float16 sQ[64 * HD_];
    __shared__ _Float16 sK[64 * HD_];
    __shared__ _Float16 sV[HD_ * 64];
    int tid = threadIdx.x, lane = tid & 31, wave = tid >> 5;
    int g = lane >> 4, ln = lane & 15;
    int qt = blockIdx.x;
    int bh = blockIdx.y, b = bh / NH_, h = bh % NH_, hk = h / NREP_;
    const _Float16* Qg  = Q + (((size_t)b * NH_  + h ) * S_ + qt * 64) * HD_;
    const _Float16* Kg0 = K + (((size_t)b * NKV_ + hk) * S_) * HD_;
    const _Float16* Vg0 = V + (((size_t)b * NKV_ + hk) * S_) * HD_;

    for (int c = tid; c < 64 * HD_ / 8; c += 128)
        async_copy16(Qg + c * 8, sQ + c * 8);

    float mrow[8], lrow[8];
    v8f zf = {};
    v8f oacc[8];
#pragma unroll
    for (int r = 0; r < 8; ++r) { mrow[r] = -1e30f; lrow[r] = 0.f; oacc[r] = zf; }

    const float smul = 0.08838834764831845f;

    for (int kt = 0; kt <= qt; ++kt) {
        __syncthreads();
        const _Float16* Kg = Kg0 + (size_t)kt * 64 * HD_;
        const _Float16* Vg = Vg0 + (size_t)kt * 64 * HD_;
        for (int c = tid; c < 1024; c += 128)
            async_copy16(Kg + c * 8, sK + c * 8);
        for (int c = tid; c < 1024; c += 128) {
            Pack8H pk; pk.v = ((const int4*)Vg)[c];
            int e0 = c * 8, key = e0 >> 7, d0 = e0 & 127;
#pragma unroll
            for (int j = 0; j < 8; ++j) sV[(d0 + j) * 64 + key] = pk.h[j];
        }
        ASYNC_WAIT(0);
        __syncthreads();

        FragH aq[4];
#pragma unroll
        for (int dc = 0; dc < 4; ++dc)
#pragma unroll
            for (int j = 0; j < 8; ++j) {
                int k = dc * 32 + g * 16 + 2 * j;
                aq[dc].u[j] = *(const unsigned int*)&sQ[(wave * 16 + ln) * HD_ + k];
            }

        v8f sc[4];
#pragma unroll
        for (int t = 0; t < 4; ++t) {
            v8f acc = zf;
#pragma unroll
            for (int dc = 0; dc < 4; ++dc) {
                FragH bk;
#pragma unroll
                for (int j = 0; j < 8; ++j) {
                    int k = dc * 32 + g * 16 + 2 * j;
                    bk.u[j] = *(const unsigned int*)&sK[(t * 16 + ln) * HD_ + k];
                }
                acc = __builtin_amdgcn_wmma_f32_16x16x32_f16(
                          false, aq[dc].v, false, bk.v, (short)0, acc, false, false);
            }
            sc[t] = acc;
        }

        int qrow0 = qt * 64 + wave * 16 + g * 8;
#pragma unroll
        for (int t = 0; t < 4; ++t)
#pragma unroll
            for (int r = 0; r < 8; ++r) {
                int kcol = kt * 64 + t * 16 + ln;
                float v = sc[t][r] * smul;
                sc[t][r] = (kcol <= qrow0 + r) ? v : -1e30f;
            }

        float mnew[8], alpha[8];
#pragma unroll
        for (int r = 0; r < 8; ++r) {
            float v = fmaxf(fmaxf(sc[0][r], sc[1][r]), fmaxf(sc[2][r], sc[3][r]));
            v = fmaxf(v, __shfl_xor(v, 1, 32));
            v = fmaxf(v, __shfl_xor(v, 2, 32));
            v = fmaxf(v, __shfl_xor(v, 4, 32));
            v = fmaxf(v, __shfl_xor(v, 8, 32));
            mnew[r]  = fmaxf(mrow[r], v);
            alpha[r] = __expf(mrow[r] - mnew[r]);
        }
#pragma unroll
        for (int t = 0; t < 4; ++t)
#pragma unroll
            for (int r = 0; r < 8; ++r)
                sc[t][r] = __expf(sc[t][r] - mnew[r]);
#pragma unroll
        for (int r = 0; r < 8; ++r) {
            float sum = sc[0][r] + sc[1][r] + sc[2][r] + sc[3][r];
            sum += __shfl_xor(sum, 1, 32);
            sum += __shfl_xor(sum, 2, 32);
            sum += __shfl_xor(sum, 4, 32);
            sum += __shfl_xor(sum, 8, 32);
            lrow[r] = lrow[r] * alpha[r] + sum;
            mrow[r] = mnew[r];
        }
#pragma unroll
        for (int dc = 0; dc < 8; ++dc)
#pragma unroll
            for (int r = 0; r < 8; ++r) oacc[dc][r] *= alpha[r];

        __syncthreads();
#pragma unroll
        for (int t = 0; t < 4; ++t)
#pragma unroll
            for (int r = 0; r < 8; ++r)
                sK[(wave * 16 + r + g * 8) * 64 + t * 16 + ln] = (_Float16)sc[t][r];
        __syncthreads();

        FragH ap[2];
#pragma unroll
        for (int kc = 0; kc < 2; ++kc)
#pragma unroll
            for (int j = 0; j < 8; ++j) {
                int k = kc * 32 + g * 16 + 2 * j;
                ap[kc].u[j] = *(const unsigned int*)&sK[(wave * 16 + ln) * 64 + k];
            }
#pragma unroll
        for (int dc = 0; dc < 8; ++dc) {
#pragma unroll
            for (int kc = 0; kc < 2; ++kc) {
                FragH bv;
#pragma unroll
                for (int j = 0; j < 8; ++j) {
                    int k = kc * 32 + g * 16 + 2 * j;
                    bv.u[j] = *(const unsigned int*)&sV[(dc * 16 + ln) * 64 + k];
                }
                oacc[dc] = __builtin_amdgcn_wmma_f32_16x16x32_f16(
                               false, ap[kc].v, false, bv.v, (short)0, oacc[dc], false, false);
            }
        }
    }

#pragma unroll
    for (int dc = 0; dc < 8; ++dc)
#pragma unroll
        for (int r = 0; r < 8; ++r) {
            int srow = qt * 64 + wave * 16 + r + g * 8;
            int d = dc * 16 + ln;
            AO[((size_t)(b * S_ + srow)) * H_ + h * HD_ + d] = oacc[dc][r] / lrow[r];
        }
}

extern "C" void kernel_launch(void* const* d_in, const int* in_sizes, int n_in,
                              void* d_out, int out_size, void* d_ws, size_t ws_size,
                              hipStream_t stream) {
    (void)in_sizes; (void)n_in; (void)out_size; (void)ws_size;
    const float* x    = (const float*)d_in[0];
    const float* cosT = (const float*)d_in[1];
    const float* sinT = (const float*)d_in[2];
    const float* wq   = (const float*)d_in[3];
    const float* wk   = (const float*)d_in[4];
    const float* wv   = (const float*)d_in[5];
    const float* wo   = (const float*)d_in[6];
    const float* qn   = (const float*)d_in[7];
    const float* kn   = (const float*)d_in[8];
    float* out = (float*)d_out;

    char* p = (char*)d_ws;
    auto alloc = [&](size_t bytes) -> char* {
        char* r = p;
        p += (bytes + 255) & ~(size_t)255;
        return r;
    };
    const size_t MROWS = (size_t)B_ * S_;
    signed char* xq8  = (signed char*)alloc(MROWS * H_);
    float*       xs   = (float*)alloc(MROWS * 4);
    signed char* wq8  = (signed char*)alloc((size_t)NH_ * HD_ * H_);
    signed char* wk8  = (signed char*)alloc((size_t)NKV_ * HD_ * H_);
    signed char* wv8  = (signed char*)alloc((size_t)NKV_ * HD_ * H_);
    signed char* wo8  = (signed char*)alloc((size_t)H_ * NH_ * HD_);
    float*       scq  = (float*)alloc(4);
    float*       sck  = (float*)alloc(4);
    float*       scv  = (float*)alloc(4);
    float*       sco  = (float*)alloc(4);
    float*       part = (float*)alloc(256 * 4);
    float*       Qf   = (float*)alloc(MROWS * NH_  * HD_ * 4);
    float*       Kf   = (float*)alloc(MROWS * NKV_ * HD_ * 4);
    float*       Vf   = (float*)alloc(MROWS * NKV_ * HD_ * 4);
    _Float16*    qh   = (_Float16*)alloc(MROWS * NH_  * HD_ * 2);
    _Float16*    kh   = (_Float16*)alloc(MROWS * NKV_ * HD_ * 2);
    _Float16*    vh   = (_Float16*)alloc(MROWS * NKV_ * HD_ * 2);
    float*       ao   = (float*)alloc(MROWS * H_ * 4);
    signed char* aoq8 = (signed char*)alloc(MROWS * H_);
    float*       aos  = (float*)alloc(MROWS * 4);

    struct WSpec { const float* w; signed char* q; float* sc; int n; };
    WSpec ws4[4] = {
        { wq, wq8, scq, NH_  * HD_ * H_ },
        { wk, wk8, sck, NKV_ * HD_ * H_ },
        { wv, wv8, scv, NKV_ * HD_ * H_ },
        { wo, wo8, sco, H_ * NH_ * HD_ },
    };
    for (int i = 0; i < 4; ++i) {
        k_abs_partial<<<256, 256, 0, stream>>>(ws4[i].w, ws4[i].n, part);
        k_abs_finish<<<1, 256, 0, stream>>>(part, (float)ws4[i].n, ws4[i].sc);
        k_wquant<<<1024, 256, 0, stream>>>(ws4[i].w, ws4[i].sc, ws4[i].q, ws4[i].n);
    }

    k_actquant<<<(int)MROWS, 256, 0, stream>>>(x, xq8, xs, H_);

    k_gemm_i8_pipe<<<dim3(NH_  * HD_ / 64, MROWS / 64), 256, 0, stream>>>(xq8, xs, wq8, scq, Qf, (int)MROWS, NH_  * HD_, H_);
    k_gemm_i8_pipe<<<dim3(NKV_ * HD_ / 64, MROWS / 64), 256, 0, stream>>>(xq8, xs, wk8, sck, Kf, (int)MROWS, NKV_ * HD_, H_);
    k_gemm_i8_pipe<<<dim3(NKV_ * HD_ / 64, MROWS / 64), 256, 0, stream>>>(xq8, xs, wv8, scv, Vf, (int)MROWS, NKV_ * HD_, H_);

    k_rmsrope<<<(int)(MROWS * NH_),  HD_, 0, stream>>>(Qf, qn, cosT, sinT, qh, NH_);
    k_rmsrope<<<(int)(MROWS * NKV_), HD_, 0, stream>>>(Kf, kn, cosT, sinT, kh, NKV_);
    k_vconv<<<2048, 256, 0, stream>>>(Vf, vh);

    k_attn<<<dim3(S_ / 64, B_ * NH_), 128, 0, stream>>>(qh, kh, vh, ao);

    k_actquant<<<(int)MROWS, 256, 0, stream>>>(ao, aoq8, aos, H_);
    k_gemm_i8_pipe<<<dim3(H_ / 64, MROWS / 64), 256, 0, stream>>>(aoq8, aos, wo8, sco, out, (int)MROWS, H_, H_);
}
